// AttentionModule_66133906423940
// MI455X (gfx1250) — matrix operand order repaired and checked by reading, not compiled, not run on hardware
//
#include <hip/hip_runtime.h>
#include <hip/hip_bf16.h>
#include <math.h>

#define SEQ   2048
#define BATCH 32
#define DIM   1024

typedef __attribute__((ext_vector_type(16))) __bf16 v16bf;
typedef __attribute__((ext_vector_type(8)))  float  v8f;
typedef __attribute__((ext_vector_type(4)))  unsigned int uint4v;
typedef __attribute__((ext_vector_type(2)))  unsigned int uint2v;
typedef __attribute__((ext_vector_type(4)))  float  float4v;

union Frag {
    uint4v q[2];
    v16bf  m;
};

__device__ __forceinline__ unsigned short f32_to_bf16_rne(float x) {
    unsigned int u = __float_as_uint(x);
    unsigned int r = u + 0x7FFFu + ((u >> 16) & 1u);
    return (unsigned short)(r >> 16);
}

// lane-xor shuffle via ds_swizzle (group-of-32 mode, and_mask=0x1f)
#define SWZ_XOR_F(x, m) \
    __int_as_float(__builtin_amdgcn_ds_swizzle(__float_as_int(x), 0x1F | ((m) << 10)))

// ---------------------------------------------------------------------------
// Kernel 0: convert W_enc (DIM x DIM fp32, row-major) -> bf16 in workspace
// ---------------------------------------------------------------------------
__global__ __launch_bounds__(256) void convert_wenc_kernel(
    const float* __restrict__ w, unsigned short* __restrict__ out) {
    int i4 = blockIdx.x * 256 + threadIdx.x;        // 0 .. 262143 (DIM*DIM/4)
    float4v f = ((const float4v*)w)[i4];
    unsigned int lo = (unsigned int)f32_to_bf16_rne(f.x) |
                      ((unsigned int)f32_to_bf16_rne(f.y) << 16);
    unsigned int hi = (unsigned int)f32_to_bf16_rne(f.z) |
                      ((unsigned int)f32_to_bf16_rne(f.w) << 16);
    uint2v p; p.x = lo; p.y = hi;
    ((uint2v*)out)[i4] = p;
}

// ---------------------------------------------------------------------------
// Kernel 1: query[b][e] = sum_d dec[b][d] * W_dec[e][d]   (tiny GEMV)
// grid (4, 32), 256 threads
// ---------------------------------------------------------------------------
__global__ __launch_bounds__(256) void query_kernel(
    const float* __restrict__ dec, const float* __restrict__ wdec,
    float* __restrict__ query) {
    __shared__ float ds[DIM];
    const int b = blockIdx.y;
    const int e = blockIdx.x * 256 + threadIdx.x;
    for (int i = threadIdx.x; i < DIM; i += 256) ds[i] = dec[b * DIM + i];
    __syncthreads();
    const float* wr = wdec + (size_t)e * DIM;
    float acc = 0.f;
    for (int d = 0; d < DIM; d += 4) {
        float4v w = *(const float4v*)(wr + d);
        acc += w.x * ds[d] + w.y * ds[d + 1] + w.z * ds[d + 2] + w.w * ds[d + 3];
    }
    query[b * DIM + e] = acc;
}

// ---------------------------------------------------------------------------
// Kernel 2: fused keys-GEMM + tanh + dot(V_att) -> logits[b][s]
// One workgroup = (batch b, 64 seq rows). 256 threads = 8 waves.
// Wave w covers e-tiles [w*8, w*8+8). Each B fragment (L2) is reused by
// 4 M-tiles (64 seq rows) -> L2 B-traffic cut 4x vs M=16 blocking.
// Per wave: 8 et * 32 k * 4 m = 1024 x v_wmma_f32_16x16x32_bf16.
// ---------------------------------------------------------------------------
#define LDS_STRIDE (DIM + 8)   // halves; +16B row pad to avoid bank conflicts
#define MROWS 64               // seq rows per workgroup (4 M-tiles)

__global__ __launch_bounds__(256) void logits_kernel(
    const float* __restrict__ enc,               // (SEQ, BATCH, DIM) fp32
    const unsigned short* __restrict__ wenc_bf,  // (DIM, DIM) bf16
    const float* __restrict__ query,             // (BATCH, DIM)
    const float* __restrict__ vatt,              // (DIM)
    float* __restrict__ logits) {                // (BATCH, SEQ)
    __shared__ unsigned short As[MROWS * LDS_STRIDE];   // 129 KB + pad
    __shared__ float lsum[MROWS];

    const int tid  = threadIdx.x;
    const int wave = tid >> 5;                   // 0..7
    const int lane = tid & 31;
    const int b     = blockIdx.x & (BATCH - 1);
    const int stile = blockIdx.x >> 5;           // 0..31
    const int s0    = stile * MROWS;

    if (tid < MROWS) lsum[tid] = 0.f;

    // Stage A tile: 64 rows x 1024 fp32 -> bf16 in LDS (each row contiguous)
    for (int i = tid; i < MROWS * (DIM / 4); i += 256) {
        const int row = i >> 8;                  // / 256 float4 per row
        const int c4  = i & 255;
        float4v f = ((const float4v*)(enc + ((size_t)(s0 + row) * BATCH + b) * DIM))[c4];
        unsigned int lo = (unsigned int)f32_to_bf16_rne(f.x) |
                          ((unsigned int)f32_to_bf16_rne(f.y) << 16);
        unsigned int hi = (unsigned int)f32_to_bf16_rne(f.z) |
                          ((unsigned int)f32_to_bf16_rne(f.w) << 16);
        uint2v p; p.x = lo; p.y = hi;
        *(uint2v*)(&As[row * LDS_STRIDE + c4 * 4]) = p;
    }
    __syncthreads();

    const int sub = lane >> 4;                   // lane half
    const int n   = lane & 15;                   // N (col) / M-in-tile index
    // A: lane holds row M = n (of its M-tile); halves 0..7 -> K = k+sub*8..+7,
    //    halves 8..15 -> K = k+16+sub*8..+7     (ISA 16-bit A layout)
    const unsigned short* arow = As + n * LDS_STRIDE + sub * 8;

    float part[4][8];
#pragma unroll
    for (int m = 0; m < 4; ++m)
#pragma unroll
        for (int j = 0; j < 8; ++j) part[m][j] = 0.f;

    for (int et = wave * 8; et < wave * 8 + 8; ++et) {
        const int e0 = et * 16;
        // B: lane holds col N = n, K = k+sub*8..+7 and k+16+sub*8..+7 (same as A)
        const unsigned short* wrow = wenc_bf + (size_t)(e0 + n) * DIM + sub * 8;

        v8f c[4];
#pragma unroll
        for (int m = 0; m < 4; ++m) c[m] = (v8f){};

#pragma unroll 4
        for (int k = 0; k < DIM; k += 32) {
            Frag bm;
            bm.q[0] = *(const uint4v*)(wrow + k);
            bm.q[1] = *(const uint4v*)(wrow + k + 16);
#pragma unroll
            for (int m = 0; m < 4; ++m) {
                Frag a;
                const unsigned short* ar = arow + (size_t)m * 16 * LDS_STRIDE + k;
                a.q[0] = *(const uint4v*)(ar);
                a.q[1] = *(const uint4v*)(ar + 16);
                c[m] = __builtin_amdgcn_wmma_f32_16x16x32_bf16(
                           false, a.m, false, bm.m, (short)0, c[m], false, false);
            }
        }
        // Epilogue: lane holds keys[s0 + m*16 + j + 8*sub][e0 + n] in c[m][j]
        const float q  = query[b * DIM + e0 + n];
        const float vv = vatt[e0 + n];
#pragma unroll
        for (int m = 0; m < 4; ++m)
#pragma unroll
            for (int j = 0; j < 8; ++j) part[m][j] += tanhf(c[m][j] + q) * vv;
    }

    // Reduce over N (16 lanes per half) via ds_swizzle butterfly
#pragma unroll
    for (int m = 0; m < 4; ++m) {
#pragma unroll
        for (int j = 0; j < 8; ++j) {
            float x = part[m][j];
            x += SWZ_XOR_F(x, 1);
            x += SWZ_XOR_F(x, 2);
            x += SWZ_XOR_F(x, 4);
            x += SWZ_XOR_F(x, 8);
            part[m][j] = x;
        }
    }
    if (n == 0) {
        const int rbase = sub * 8;               // lanes 0/16 own rows j / j+8
#pragma unroll
        for (int m = 0; m < 4; ++m)
#pragma unroll
            for (int j = 0; j < 8; ++j)
                atomicAdd(&lsum[m * 16 + rbase + j], part[m][j]);
    }
    __syncthreads();
    if (tid < MROWS) logits[b * SEQ + s0 + tid] = lsum[tid];
}

// ---------------------------------------------------------------------------
// Kernel 3: softmax over seq per batch -> alpha (written to d_out region)
// ---------------------------------------------------------------------------
__global__ __launch_bounds__(256) void softmax_kernel(
    const float* __restrict__ logits, float* __restrict__ alpha) {
    __shared__ float red[256];
    const int b = blockIdx.x, tid = threadIdx.x;
    float l[8];
    float m = -INFINITY;
#pragma unroll
    for (int i = 0; i < 8; ++i) {
        l[i] = logits[b * SEQ + tid + i * 256];
        m = fmaxf(m, l[i]);
    }
    red[tid] = m; __syncthreads();
    for (int s = 128; s > 0; s >>= 1) {
        if (tid < s) red[tid] = fmaxf(red[tid], red[tid + s]);
        __syncthreads();
    }
    m = red[0]; __syncthreads();
    float sum = 0.f;
#pragma unroll
    for (int i = 0; i < 8; ++i) { l[i] = __expf(l[i] - m); sum += l[i]; }
    red[tid] = sum; __syncthreads();
    for (int s = 128; s > 0; s >>= 1) {
        if (tid < s) red[tid] += red[tid + s];
        __syncthreads();
    }
    const float inv = 1.0f / red[0];
#pragma unroll
    for (int i = 0; i < 8; ++i) alpha[b * SEQ + tid + i * 256] = l[i] * inv;
}

// ---------------------------------------------------------------------------
// Kernel 4: context partials: grid (dchunk=4, schunk=4, b=32), 256 threads
// ---------------------------------------------------------------------------
__global__ __launch_bounds__(256) void context_partial_kernel(
    const float* __restrict__ enc, const float* __restrict__ alpha,
    float* __restrict__ partials) {
    __shared__ float al[SEQ / 4];
    const int d  = blockIdx.x * 256 + threadIdx.x;
    const int sc = blockIdx.y;
    const int b  = blockIdx.z;
    const int sbeg = sc * (SEQ / 4);
    for (int i = threadIdx.x; i < SEQ / 4; i += 256)
        al[i] = alpha[b * SEQ + sbeg + i];
    __syncthreads();
    float acc = 0.f;
    for (int s = 0; s < SEQ / 4; ++s)
        acc += al[s] * enc[((size_t)(sbeg + s) * BATCH + b) * DIM + d];
    partials[((b * 4 + sc) * DIM) + d] = acc;
}

__global__ __launch_bounds__(256) void context_reduce_kernel(
    const float* __restrict__ partials, float* __restrict__ ctx) {
    const int i = blockIdx.x * 256 + threadIdx.x;    // b*DIM + d
    const int b = i >> 10, d = i & (DIM - 1);
    float s = 0.f;
#pragma unroll
    for (int sc = 0; sc < 4; ++sc) s += partials[((b * 4 + sc) * DIM) + d];
    ctx[i] = s;
}

// ---------------------------------------------------------------------------
extern "C" void kernel_launch(void* const* d_in, const int* in_sizes, int n_in,
                              void* d_out, int out_size, void* d_ws, size_t ws_size,
                              hipStream_t stream) {
    const float* dec  = (const float*)d_in[0];   // (32,1,1024)
    const float* enc  = (const float*)d_in[1];   // (2048,32,1024)
    const float* wenc = (const float*)d_in[2];   // (1024,1024)
    const float* wdec = (const float*)d_in[3];   // (1024,1024)
    const float* vatt = (const float*)d_in[4];   // (1,1024)

    float* out   = (float*)d_out;
    float* ctx   = out;                          // 32*1024 floats
    float* alpha = out + BATCH * DIM;            // 32*2048 floats, (1,B,S)

    char* ws = (char*)d_ws;
    unsigned short* wenc_bf = (unsigned short*)ws;                       // 2 MB
    float* query    = (float*)(ws + (size_t)(2u << 20));                 // 128 KB
    float* logits   = (float*)(ws + (size_t)(2u << 20) + (128u << 10));  // 256 KB
    float* partials = (float*)(ws + (size_t)(2u << 20) + (384u << 10));  // 512 KB

    convert_wenc_kernel<<<DIM * DIM / (256 * 4), 256, 0, stream>>>(wenc, wenc_bf);
    query_kernel<<<dim3(DIM / 256, BATCH), 256, 0, stream>>>(dec, wdec, query);
    logits_kernel<<<BATCH * (SEQ / MROWS), 256, 0, stream>>>(enc, wenc_bf, query, vatt, logits);
    softmax_kernel<<<BATCH, 256, 0, stream>>>(logits, alpha);
    context_partial_kernel<<<dim3(DIM / 256, 4, BATCH), 256, 0, stream>>>(enc, alpha, partials);
    context_reduce_kernel<<<BATCH * DIM / 256, 256, 0, stream>>>(partials, ctx);
}
